// MooreDeepVanillaRNN_23356032155739
// MI455X (gfx1250) — compile-verified, matrix operand order checked by reading, not run on hardware
//
#include <hip/hip_runtime.h>
#include <hip/hip_bf16.h>
#include <math.h>

// ---------------------------------------------------------------------------
// MooreDeepVanillaRNN on MI455X (gfx1250, wave32, WMMA + TDM).
//
// 9 persistent workgroups = 8 RNN layer stages + 1 head stage, software-
// pipelined over timesteps. Each layer stage keeps its bf16 weights
// transposed ([N=256][K=512], 256KB) in LDS plus 32KB input + 32KB recurrent
// hidden tiles (320KB = full WGP LDS). Per timestep each stage does a
// [64,512]x[512,256] GEMM with v_wmma_f32_16x16x32_bf16 (f32 accumulate),
// bias+tanh, and hands the bf16 [64,256] activation to the next stage through
// double-buffered L2-resident global slots guarded by acquire/release
// counters with backpressure. Inter-stage tiles are pulled into LDS by the
// Tensor Data Mover (tensor_load_to_lds + s_wait_tensorcnt) when available,
// freeing the SIMDs of all copy traffic. Head stage does [64,256]x[256,128]
// WMMA and writes fp32 logits to d_out.
// ---------------------------------------------------------------------------

#define B_     64
#define T_     512
#define D_     256
#define K2_    512   // 2*D
#define L_     8
#define NS_    128
#define NTHR   512   // 16 waves of 32

typedef __attribute__((ext_vector_type(16))) __bf16 v16bf;
typedef __attribute__((ext_vector_type(8)))  float  v8f;
typedef __attribute__((ext_vector_type(4)))  unsigned int v4u;
typedef __attribute__((ext_vector_type(8)))  int    v8i_;
typedef __attribute__((ext_vector_type(4)))  int    v4i_;

union FragU { v16bf v; uint4 q[2]; };

// Load one 16-bit WMMA fragment row for this lane: two contiguous 16-byte
// chunks at kk and kk+16 (ISA 7.12.2 layout: lane half selects the k-phase,
// which the caller folds into kk).
__device__ __forceinline__ v16bf ld_frag(const __bf16* rowbase, int kk) {
  FragU f;
  f.q[0] = *(const uint4*)(rowbase + kk);
  f.q[1] = *(const uint4*)(rowbase + kk + 16);
  return f.v;
}

__device__ __forceinline__ v8f wmma_bf16(v16bf a, v16bf b, v8f c) {
  return __builtin_amdgcn_wmma_f32_16x16x32_bf16(
      /*neg_a=*/false, a, /*neg_b=*/false, b,
      /*c_mod=*/(short)0, c, /*reuse_a=*/false, /*reuse_b=*/false);
}

__device__ __forceinline__ void spin_until_ge(int* p, int val) {
  while (__hip_atomic_load(p, __ATOMIC_ACQUIRE, __HIP_MEMORY_SCOPE_AGENT) < val)
    __builtin_amdgcn_s_sleep(1);
}

__device__ __forceinline__ void wait_tensorcnt0() {
#if __has_builtin(__builtin_amdgcn_s_wait_tensorcnt)
  __builtin_amdgcn_s_wait_tensorcnt(0);
#else
  asm volatile("s_wait_tensorcnt 0x0" ::: "memory");
#endif
}

#if __has_builtin(__builtin_amdgcn_tensor_load_to_lds)
#define HAVE_TDM 1
// DMA a contiguous 32KB bf16 tile (16384 elements) from global memory into
// LDS via the Tensor Data Mover. 2D descriptor per ISA 8.3/8.4:
//   group0: count=1, lds_addr, 57-bit global_addr, type=2 ("image")
//   group1: data_size=1 (2B), dim0=tile_dim0=16384, dim1=tile_dim1=1,
//           dim0_stride=16384, no multicast/pad/iterate/atomic-barrier
//   groups 2/3 (+ extra descriptor words on this toolchain): zero (<=2D).
// Issued by a single wave; tracked with TENSORcnt (EXEC is ignored).
__device__ __forceinline__ void tdm_load_tile_32k(const void* gsrc, void* lds_dst) {
  unsigned long long ga = (unsigned long long)(uintptr_t)gsrc;
  // Flat LDS pointer: addr[31:0] is the LDS byte offset (aperture rule).
  unsigned int lds_addr = (unsigned int)(uintptr_t)lds_dst;
  const unsigned int elems = B_ * D_;            // 16384 bf16 elements
  v4u g0;
  g0[0] = 1u;                                    // count=1, user descriptor
  g0[1] = lds_addr;                              // lds_addr [63:32]
  g0[2] = (unsigned int)(ga & 0xFFFFFFFFu);      // global_addr lo [95:64]
  g0[3] = (unsigned int)((ga >> 32) & 0x01FFFFFFu) | (2u << 30); // addr hi, type=2
  v8i_ g1;
  g1[0] = (int)(1u << 16);                       // wg_mask=0, data_size=1 (2B)
  g1[1] = (int)((elems & 0xFFFFu) << 16);        // tensor_dim0 lo16 @ [63:48]
  g1[2] = (int)(((elems >> 16) & 0xFFFFu) | (1u << 16)); // dim0 hi16, tensor_dim1=1
  g1[3] = (int)((elems & 0xFFFFu) << 16);        // dim1 hi16=0, tile_dim0
  g1[4] = 1;                                     // tile_dim1=1, tile_dim2=0
  g1[5] = (int)elems;                            // tensor_dim0_stride lo32
  g1[6] = 0;                                     // stride hi, dim1_stride lo
  g1[7] = 0;                                     // dim1_stride hi
  v4i_ gz  = {0, 0, 0, 0};                       // groups 2/3: unused (2D)
  v8i_ gz8 = {0, 0, 0, 0, 0, 0, 0, 0};           // extra words (6-arg form)
  // This toolchain exposes the 6-arg builtin:
  // (v4u g0, v8i g1, v4i g2, v4i g3, v8i extra, i32 cpol)
  __builtin_amdgcn_tensor_load_to_lds(g0, g1, gz, gz, gz8, 0);
}
#else
#define HAVE_TDM 0
#endif

// Stage a [64][256] bf16 tile from a global slot into LDS.
__device__ __forceinline__ void stage_tile(const __bf16* gsrc, __bf16* Ain,
                                           int tid, int wv) {
#if HAVE_TDM
  if (wv == 0) {                 // one wave issues the DMA
    tdm_load_tile_32k((const void*)gsrc, (void*)Ain);
    wait_tensorcnt0();           // data in LDS before the barrier releases
  }
  (void)tid;
#else
  const uint4* src = (const uint4*)gsrc;
  uint4* dst = (uint4*)Ain;
  for (int i = tid; i < (B_ * D_) / 8; i += NTHR) dst[i] = src[i];
#endif
}

__global__ void rnn_init_flags(int* flags) {
  if (threadIdx.x < 64) flags[threadIdx.x] = 0;
}

__global__ void __launch_bounds__(NTHR)
rnn_pipeline(const int* __restrict__ ids,      // [B][T]
             const float* __restrict__ emb,    // [VOCAB][D]
             const float* __restrict__ Ws,     // [L][2D][D]
             const float* __restrict__ bs,     // [L][D]
             const float* __restrict__ head_w, // [D][NS]
             float* __restrict__ out,          // [B][T][NS]
             __bf16* __restrict__ gbuf,        // [L][2][B][D] bf16 slots
             int* __restrict__ flags)          // ready[16], consumed[16]
{
  extern __shared__ char smem[];
  const int stage = blockIdx.x;      // 0..7 layers, 8 = head
  const int tid   = threadIdx.x;
  const int lane  = tid & 31;
  const int wv    = tid >> 5;        // wave id 0..15
  const int half  = lane >> 4;       // lane group (k-phase)
  const int lr    = lane & 15;       // row/col within 16-tile
  int* ready    = flags;             // ready[l]   : #timesteps published
  int* consumed = flags + 16;        // consumed[l]: #timesteps consumed

  if (stage < L_) {
    // ---------------- layer stage ----------------
    __bf16* Wt   = (__bf16*)smem;                      // [256][512] (N-major)
    __bf16* Ain  = (__bf16*)(smem + 262144);           // [64][256] input
    __bf16* Hown = (__bf16*)(smem + 262144 + 32768);   // [64][256] hidden

    // Prologue: weights -> LDS, transposed to [n][k] so B-fragments are
    // contiguous in k; hidden state zeroed.
    const float* Wsl = Ws + (size_t)stage * (K2_ * D_);
    for (int i = tid; i < K2_ * D_; i += NTHR) {
      int k = i >> 8;       // row in [0,512)
      int n = i & 255;      // col in [0,256)
      Wt[n * K2_ + k] = (__bf16)Wsl[i];
    }
    for (int i = tid; i < B_ * D_; i += NTHR) Hown[i] = (__bf16)0.0f;
    const float bias = bs[stage * D_ + wv * 16 + lr];
    const __bf16* wrow = Wt + (wv * 16 + lr) * K2_;     // this lane's N column
    const __bf16* ar0 = Ain  + (0 * 16 + lr) * D_;
    const __bf16* ar1 = Ain  + (1 * 16 + lr) * D_;
    const __bf16* ar2 = Ain  + (2 * 16 + lr) * D_;
    const __bf16* ar3 = Ain  + (3 * 16 + lr) * D_;
    const __bf16* hr0 = Hown + (0 * 16 + lr) * D_;
    const __bf16* hr1 = Hown + (1 * 16 + lr) * D_;
    const __bf16* hr2 = Hown + (2 * 16 + lr) * D_;
    const __bf16* hr3 = Hown + (3 * 16 + lr) * D_;
    __syncthreads();

    for (int t = 0; t < T_; ++t) {
      // Wait for upstream data + downstream slot free (double buffered).
      if (tid == 0) {
        if (stage > 0) spin_until_ge(&ready[stage - 1], t + 1);
        if (t >= 2)    spin_until_ge(&consumed[stage], t - 1);
      }
      __syncthreads();

      // Stage the layer input into LDS.
      if (stage == 0) {
        // Fused embedding gather + bf16 cast.
        for (int i = tid; i < B_ * D_; i += NTHR) {
          int b = i >> 8, dd = i & 255;
          int tok = ids[b * T_ + t];
          Ain[i] = (__bf16)emb[(size_t)tok * D_ + dd];
        }
      } else {
        stage_tile(gbuf + (size_t)((stage - 1) * 2 + (t & 1)) * (B_ * D_),
                   Ain, tid, wv);
      }
      __syncthreads();
      if (tid == 0 && stage > 0)
        __hip_atomic_fetch_add(&consumed[stage - 1], 1, __ATOMIC_RELEASE,
                               __HIP_MEMORY_SCOPE_AGENT);

      // GEMM: wave wv owns N column wv*16, all four M tiles. 16 k-steps of 32.
      v8f acc0 = {}, acc1 = {}, acc2 = {}, acc3 = {};
#pragma unroll
      for (int ks = 0; ks < 8; ++ks) {          // K 0..255: input half
        int kk = ks * 32 + half * 8;
        v16bf bf = ld_frag(wrow, kk);
        acc0 = wmma_bf16(ld_frag(ar0, kk), bf, acc0);
        acc1 = wmma_bf16(ld_frag(ar1, kk), bf, acc1);
        acc2 = wmma_bf16(ld_frag(ar2, kk), bf, acc2);
        acc3 = wmma_bf16(ld_frag(ar3, kk), bf, acc3);
      }
#pragma unroll
      for (int ks = 0; ks < 8; ++ks) {          // K 256..511: recurrent half
        int kk = ks * 32 + half * 8;
        v16bf bf = ld_frag(wrow, 256 + kk);
        acc0 = wmma_bf16(ld_frag(hr0, kk), bf, acc0);
        acc1 = wmma_bf16(ld_frag(hr1, kk), bf, acc1);
        acc2 = wmma_bf16(ld_frag(hr2, kk), bf, acc2);
        acc3 = wmma_bf16(ld_frag(hr3, kk), bf, acc3);
      }
      __syncthreads();  // everyone done reading Hown/Ain before we rewrite

      // Epilogue: bias + tanh, update hidden in LDS, publish bf16 downstream.
      __bf16* gout = gbuf + (size_t)(stage * 2 + (t & 1)) * (B_ * D_);
      const int ncol = wv * 16 + lr;
#define EPI(ACC, MT)                                                   \
      {                                                                \
        _Pragma("unroll")                                              \
        for (int r = 0; r < 8; ++r) {                                  \
          int m = (MT) * 16 + half * 8 + r;   /* C layout: 7.12.2 */   \
          float v = tanhf(ACC[r] + bias);                              \
          __bf16 hv = (__bf16)v;                                       \
          Hown[m * D_ + ncol] = hv;                                    \
          gout[m * D_ + ncol] = hv;                                    \
        }                                                              \
      }
      EPI(acc0, 0) EPI(acc1, 1) EPI(acc2, 2) EPI(acc3, 3)
#undef EPI
      __threadfence();   // make our stores device-visible (release)
      __syncthreads();
      if (tid == 0)
        __hip_atomic_fetch_add(&ready[stage], 1, __ATOMIC_RELEASE,
                               __HIP_MEMORY_SCOPE_AGENT);
    }
  } else {
    // ---------------- head stage: logits = h7 @ head_w ----------------
    __bf16* WtH = (__bf16*)smem;             // [128][256] (N-major)
    __bf16* Ain = (__bf16*)(smem + 65536);   // [64][256]
    for (int i = tid; i < D_ * NS_; i += NTHR) {
      int k = i >> 7, n = i & 127;           // head_w is [256][128]
      WtH[n * D_ + k] = (__bf16)head_w[i];
    }
    const int nt = wv & 7;                   // N tile 0..7
    const int mg = wv >> 3;                  // M tile group 0..1 (2 tiles each)
    const __bf16* wrow = WtH + (nt * 16 + lr) * D_;
    const __bf16* ar0 = Ain + ((mg * 2 + 0) * 16 + lr) * D_;
    const __bf16* ar1 = Ain + ((mg * 2 + 1) * 16 + lr) * D_;
    __syncthreads();

    for (int t = 0; t < T_; ++t) {
      if (tid == 0) spin_until_ge(&ready[L_ - 1], t + 1);
      __syncthreads();
      stage_tile(gbuf + (size_t)((L_ - 1) * 2 + (t & 1)) * (B_ * D_),
                 Ain, tid, wv);
      __syncthreads();
      if (tid == 0)
        __hip_atomic_fetch_add(&consumed[L_ - 1], 1, __ATOMIC_RELEASE,
                               __HIP_MEMORY_SCOPE_AGENT);

      v8f acc0 = {}, acc1 = {};
#pragma unroll
      for (int ks = 0; ks < 8; ++ks) {       // K = 256
        int kk = ks * 32 + half * 8;
        v16bf bf = ld_frag(wrow, kk);
        acc0 = wmma_bf16(ld_frag(ar0, kk), bf, acc0);
        acc1 = wmma_bf16(ld_frag(ar1, kk), bf, acc1);
      }
      const int ncol = nt * 16 + lr;
#pragma unroll
      for (int r = 0; r < 8; ++r) {
        int m0 = (mg * 2 + 0) * 16 + half * 8 + r;
        int m1 = (mg * 2 + 1) * 16 + half * 8 + r;
        out[((size_t)m0 * T_ + t) * NS_ + ncol] = acc0[r];
        out[((size_t)m1 * T_ + t) * NS_ + ncol] = acc1[r];
      }
      __syncthreads();  // don't overwrite Ain while others still read it
    }
  }
}

extern "C" void kernel_launch(void* const* d_in, const int* in_sizes, int n_in,
                              void* d_out, int out_size, void* d_ws, size_t ws_size,
                              hipStream_t stream) {
  const int*   ids    = (const int*)d_in[0];
  const float* emb    = (const float*)d_in[1];
  const float* Ws     = (const float*)d_in[2];
  const float* bs     = (const float*)d_in[3];
  const float* head_w = (const float*)d_in[4];
  float* out = (float*)d_out;

  // Workspace: [0,64) ints = ready/consumed counters; then 8 stages x 2 slots
  // of [64][256] bf16 (32KB each) => ~513KB total.
  int*    flags = (int*)d_ws;
  __bf16* gbuf  = (__bf16*)((char*)d_ws + 1024);

  rnn_init_flags<<<1, 64, 0, stream>>>(flags);

  // 9 persistent stages, 16 waves each, full 320KB LDS per workgroup
  // (256KB bf16 weights + 2x32KB activation tiles for the layer stages).
  const size_t smem_bytes = 320 * 1024;
  rnn_pipeline<<<9, NTHR, smem_bytes, stream>>>(ids, emb, Ws, bs, head_w,
                                                out, gbuf, flags);
}
